// GotenNet_70901320122385
// MI455X (gfx1250) — matrix operand order repaired and checked by reading, not compiled, not run on hardware
//
#include <hip/hip_runtime.h>
#include <math.h>

#define NN 20000
#define EE 320000
#define FF 128
#define HH 8
#define F3 384

typedef __bf16 bf16_t;
typedef __attribute__((ext_vector_type(16))) __bf16 bf16x16;
typedef __attribute__((ext_vector_type(8)))  float  f32x8;

static constexpr float RCUT       = 5.0f;
static constexpr float PI_OVER_RC = 0.6283185307179586f;   // pi / 5
static constexpr float INV_SQRT_F = 0.08838834764831845f;  // 1/sqrt(128)

// bf16 weight-fragment pool offsets (elements). Each FxF matrix = 16384,
// each 3FxF matrix = 49152 fragment elements.
static constexpr size_t OFF_QW   = 0;
static constexpr size_t OFF_KW   = 16384;
static constexpr size_t OFF_GS1  = 32768;
static constexpr size_t OFF_GS2  = 49152;
static constexpr size_t OFF_GV1  = 98304;
static constexpr size_t OFF_GV2  = 114688;
static constexpr size_t OFF_PHIK = 163840;
static constexpr size_t OFF_WRE  = 180224;
static constexpr size_t OFF_EUP  = 229376;
static constexpr size_t OFF_VECQ = 245760;
static constexpr size_t OFF_VECK = 262144;
static constexpr size_t WB_ELEMS = 278528;

__device__ __forceinline__ bf16_t to_bf16(float f) {
  unsigned u = __float_as_uint(f);
  unsigned r = u + 0x7FFFu + ((u >> 16) & 1u);   // round-to-nearest-even
  unsigned short h = (unsigned short)(r >> 16);
  bf16_t out;
  __builtin_memcpy(&out, &h, sizeof(h));
  return out;
}

__device__ __forceinline__ float silu(float v) { return v / (1.0f + __expf(-v)); }

__device__ __forceinline__ void atomic_max_f32(float* addr, float val) {
  int* ai = reinterpret_cast<int*>(addr);
  int old = __float_as_int(*addr);
  while (__int_as_float(old) < val) {
    int assumed = old;
    old = atomicCAS(ai, assumed, __float_as_int(val));
    if (old == assumed) break;
  }
}

// A-matrix (16x32 bf16) intra-tile K mapping for element i of a lane.
__device__ __forceinline__ int afrag_K(int i, int half) {
  int p = i >> 1, lohi = i & 1, grp = p >> 2, pp = p & 3;
  return grp * 16 + half * 8 + pp * 2 + lohi;
}

// Stage a 16 x 128 f32 tile (row stride FF) into LDS as WMMA bf16 A-fragments,
// slot order = (ktile, lane, elem) per the ISA 16-bit A-matrix 16x32 layout.
__device__ __forceinline__ void stage_A(const float* __restrict__ src, int row0,
                                        bf16_t* __restrict__ lds, int tid) {
#pragma unroll
  for (int ts = 0; ts < 8; ++ts) {
    int slot = tid + ts * 256;
    int i    = slot & 15;
    int lane = (slot >> 4) & 31;
    int kt   = slot >> 9;            // 0..3 (K blocks of 32)
    int mrow = lane & 15;
    int half = lane >> 4;
    int K = kt * 32 + afrag_K(i, half);
    lds[slot] = to_bf16(src[(size_t)(row0 + mrow) * FF + K]);
  }
}

// CDNA5 async copy: bring one pre-packed 4KB bf16 A-fragment tile from global
// into LDS without touching VGPR data. One global_load_async_to_lds_b128 per
// wave (32 lanes x 16B); tracked with ASYNCcnt.
__device__ __forceinline__ void stage_async_frag(const bf16_t* __restrict__ gsrc,
                                                 bf16_t* __restrict__ lds, int tid) {
  unsigned ldsbyte = (unsigned)(size_t)lds + (unsigned)(tid * 16);
  unsigned goff    = (unsigned)(tid * 16);
  unsigned long long gbase = (unsigned long long)(size_t)gsrc;
  asm volatile("global_load_async_to_lds_b128 %0, %1, %2"
               :: "v"(ldsbyte), "v"(goff), "s"(gbase)
               : "memory");
}

__device__ __forceinline__ void wait_async0() {
  asm volatile("s_wait_asynccnt 0" ::: "memory");
}

// One 16x16 f32 tile: A fragments from LDS, B fragments pre-swizzled in global,
// K = 128 via 4 chained v_wmma_f32_16x16x32_bf16.
__device__ __forceinline__ f32x8 gemm_tile(const bf16_t* __restrict__ ldsA,
                                           const bf16_t* __restrict__ wfrag,
                                           int ct, int lane) {
  f32x8 acc = {};
#pragma unroll
  for (int kt = 0; kt < 4; ++kt) {
    bf16x16 a = *reinterpret_cast<const bf16x16*>(ldsA + (size_t)(kt * 32 + lane) * 16);
    bf16x16 b = *reinterpret_cast<const bf16x16*>(
        wfrag + (size_t)((ct * 4 + kt) * 32 + lane) * 16);
    acc = __builtin_amdgcn_wmma_f32_16x16x32_bf16(false, a, false, b,
                                                  (short)0, acc, false, false);
  }
  return acc;
}

__device__ __forceinline__ void store_tile(float* __restrict__ dst, int row0,
                                           int stride, int c0, int lane,
                                           const f32x8& acc) {
  int n = lane & 15, half = lane >> 4;
#pragma unroll
  for (int r = 0; r < 8; ++r)
    dst[(size_t)(row0 + r + 8 * half) * stride + c0 + n] = acc[r];
}

// Scatter a post-activation 16x16 tile into LDS laid out as the A-fragment of
// the next GEMM (its K index = ct*16 + n). Inverse of the A-matrix K mapping.
__device__ __forceinline__ void store_Afrag(bf16_t* __restrict__ lds, int ct,
                                            int lane, const f32x8& v) {
  int n = lane & 15, half = lane >> 4;
  int Kg = ct * 16 + n;
  int kt = Kg >> 5, Kp = Kg & 31;
  int grp = Kp >> 4, rem = Kp & 15;
  int halfA = rem >> 3, pp = (rem & 7) >> 1, lohi = rem & 1;
  int i = grp * 8 + pp * 2 + lohi;
#pragma unroll
  for (int r = 0; r < 8; ++r) {
    int mrow = r + 8 * half;
    int laneA = halfA * 16 + mrow;
    lds[(size_t)(kt * 32 + laneA) * 16 + i] = to_bf16(v[r]);
  }
}

// ---------------------------------------------------------------------------
// Prep: pack one (Cout x 128) f32 weight into bf16 B-fragment order:
// idx = (((ct*4 + kt)*32 + lane)*16 + i); B elem (K = kt*32 + afrag_K(i, half),
// N = ct*16 + n) = W[N*F + K].
__global__ void __launch_bounds__(256) swizzle_w_kernel(
    const float* __restrict__ W, bf16_t* __restrict__ out, int total) {
  int idx = blockIdx.x * 256 + threadIdx.x;
  if (idx >= total) return;
  int i    = idx & 15;
  int lane = (idx >> 4) & 31;
  int kt   = (idx >> 9) & 3;
  int ct   = idx >> 11;
  int n = lane & 15, half = lane >> 4;
  int K = kt * 32 + afrag_K(i, half);
  out[idx] = to_bf16(W[(size_t)(ct * 16 + n) * FF + K]);
}

// Prep: pack r_ij (E x 128 f32) into per-16-edge-tile bf16 A-fragment order so
// edge kernels can async-DMA tiles straight into LDS.
__global__ void __launch_bounds__(256) swizzle_r_kernel(
    const float* __restrict__ r_ij, bf16_t* __restrict__ out) {
  size_t idx = (size_t)blockIdx.x * 256 + threadIdx.x;
  if (idx >= (size_t)EE * FF) return;
  int slot = (int)(idx & 2047);
  size_t bt = idx >> 11;               // 16-edge tile index
  int i    = slot & 15;
  int lane = (slot >> 4) & 31;
  int kt   = slot >> 9;
  int mrow = lane & 15;
  int half = lane >> 4;
  int K = kt * 32 + afrag_K(i, half);
  out[idx] = to_bf16(r_ij[(bt * 16 + mrow) * FF + K]);
}

// Prep: s_out = s, t_out = t (segment sums accumulate atomically on top),
// m = -inf, den = 0.
__global__ void __launch_bounds__(256) init_kernel(
    const float* __restrict__ s, const float* __restrict__ t,
    float* __restrict__ s_out, float* __restrict__ t_out,
    float* __restrict__ m, float* __restrict__ den) {
  size_t idx = (size_t)blockIdx.x * 256 + threadIdx.x;
  if (idx < (size_t)NN * FF) s_out[idx] = s[idx];
  if (idx < (size_t)NN * F3) t_out[idx] = t[idx];
  if (idx < (size_t)NN * HH) { m[idx] = -3.0e38f; den[idx] = 0.0f; }
}

// ---------------------------------------------------------------------------
// Node-side fused GEMMs: q, k, x = gs2(silu(gs1(s2))), val = gv2(silu(gv1(s2))).
__global__ void __launch_bounds__(256) node_gemm_kernel(
    const float* __restrict__ s2, const bf16_t* __restrict__ wb,
    const float* __restrict__ qb, const float* __restrict__ kb,
    const float* __restrict__ gs1b, const float* __restrict__ gs2b,
    const float* __restrict__ gv1b, const float* __restrict__ gv2b,
    float* __restrict__ q, float* __restrict__ k,
    float* __restrict__ x, float* __restrict__ val) {
  __shared__ __align__(64) bf16_t ldsA[2048];
  __shared__ __align__(64) bf16_t ldsH[2048];
  const int tid = threadIdx.x, lane = tid & 31, w = tid >> 5;
  const int n = lane & 15;
  const int row0 = blockIdx.x * 16;

  stage_A(s2, row0, ldsA, tid);
  __syncthreads();

  {
    f32x8 acc = gemm_tile(ldsA, wb + OFF_QW, w, lane);
    float b = qb[w * 16 + n];
#pragma unroll
    for (int r = 0; r < 8; ++r) acc[r] += b;
    store_tile(q, row0, FF, w * 16, lane, acc);
  }
  {
    f32x8 acc = gemm_tile(ldsA, wb + OFF_KW, w, lane);
    float b = kb[w * 16 + n];
#pragma unroll
    for (int r = 0; r < 8; ++r) acc[r] += b;
    store_tile(k, row0, FF, w * 16, lane, acc);
  }

  {  // hidden = silu(gs1(s2)) -> LDS as next A fragment
    f32x8 acc = gemm_tile(ldsA, wb + OFF_GS1, w, lane);
    float b = gs1b[w * 16 + n];
#pragma unroll
    for (int r = 0; r < 8; ++r) acc[r] = silu(acc[r] + b);
    store_Afrag(ldsH, w, lane, acc);
  }
  __syncthreads();
#pragma unroll
  for (int tt = 0; tt < 3; ++tt) {  // x: 24 col tiles over 8 waves
    int ct = w + tt * 8;
    f32x8 acc = gemm_tile(ldsH, wb + OFF_GS2, ct, lane);
    float b = gs2b[ct * 16 + n];
#pragma unroll
    for (int r = 0; r < 8; ++r) acc[r] += b;
    store_tile(x, row0, F3, ct * 16, lane, acc);
  }
  __syncthreads();

  {  // hidden = silu(gv1(s2))
    f32x8 acc = gemm_tile(ldsA, wb + OFF_GV1, w, lane);
    float b = gv1b[w * 16 + n];
#pragma unroll
    for (int r = 0; r < 8; ++r) acc[r] = silu(acc[r] + b);
    store_Afrag(ldsH, w, lane, acc);
  }
  __syncthreads();
#pragma unroll
  for (int tt = 0; tt < 3; ++tt) {  // val
    int ct = w + tt * 8;
    f32x8 acc = gemm_tile(ldsH, wb + OFF_GV2, ct, lane);
    float b = gv2b[ct * 16 + n];
#pragma unroll
    for (int r = 0; r < 8; ++r) acc[r] += b;
    store_tile(val, row0, F3, ct * 16, lane, acc);
  }
}

// ---------------------------------------------------------------------------
// Edge logits: silu(phik(r_ij)) fused with q[dst]*k[src] reduction per head
// (wave id == head since F/H == 16) + segment max via float atomic-CAS max.
__global__ void __launch_bounds__(256) edge_logits_kernel(
    const bf16_t* __restrict__ r_frag, const bf16_t* __restrict__ wb,
    const float* __restrict__ phikb, const int* __restrict__ ei,
    const float* __restrict__ q, const float* __restrict__ k,
    float* __restrict__ logits, float* __restrict__ mbuf) {
  __shared__ __align__(64) bf16_t ldsA[2048];
  const int tid = threadIdx.x, lane = tid & 31, h = tid >> 5;
  const int n = lane & 15, half = lane >> 4;
  const int e0 = blockIdx.x * 16;

  stage_async_frag(r_frag + (size_t)blockIdx.x * 2048, ldsA, tid);
  if (blockIdx.x + 1 < (int)gridDim.x)
    __builtin_prefetch(r_frag + (size_t)(blockIdx.x + 1) * 2048 + tid * 8, 0, 1);
  wait_async0();
  __syncthreads();

  f32x8 acc = gemm_tile(ldsA, wb + OFF_PHIK, h, lane);
  float b = phikb[h * 16 + n];
#pragma unroll
  for (int r = 0; r < 8; ++r) {
    int e = e0 + r + 8 * half;
    int src = ei[e], dst = ei[EE + e];
    float ra = silu(acc[r] + b);
    float p = q[(size_t)dst * FF + h * 16 + n] *
              k[(size_t)src * FF + h * 16 + n] * ra;
    p += __shfl_xor(p, 1, 32);   // reduce over 16 lanes of each half
    p += __shfl_xor(p, 2, 32);
    p += __shfl_xor(p, 4, 32);
    p += __shfl_xor(p, 8, 32);
    if (n == 0) {
      logits[(size_t)e * HH + h] = p;
      atomic_max_f32(&mbuf[(size_t)dst * HH + h], p);
    }
  }
}

// exp(logit - m[dst]) in place + segment-sum denominators.
__global__ void __launch_bounds__(256) softmax_ex_kernel(
    float* __restrict__ exbuf, const float* __restrict__ mbuf,
    float* __restrict__ den, const int* __restrict__ ei) {
  size_t idx = (size_t)blockIdx.x * 256 + threadIdx.x;
  if (idx >= (size_t)EE * HH) return;
  int e = (int)(idx / HH), h = (int)(idx % HH);
  int dst = ei[EE + e];
  float v = __expf(exbuf[idx] - mbuf[(size_t)dst * HH + h]);
  exbuf[idx] = v;
  atomicAdd(&den[(size_t)dst * HH + h], v);
}

// ---------------------------------------------------------------------------
// Main edge kernel: r_e = wre(r_ij) (WMMA), attention-weighted value, cosine
// cutoff mix, split o_s/o_d/o_t, scatter-add into s_out / t_out.
__global__ void __launch_bounds__(256) edge_main_kernel(
    const bf16_t* __restrict__ r_frag, const bf16_t* __restrict__ wb,
    const float* __restrict__ wreb, const int* __restrict__ ei,
    const float* __restrict__ exbuf, const float* __restrict__ den,
    const float* __restrict__ nee, const float* __restrict__ d_ij,
    const float* __restrict__ dirv, const float* __restrict__ x,
    const float* __restrict__ val, const float* __restrict__ t_in,
    float* __restrict__ s_out, float* __restrict__ t_out) {
  __shared__ __align__(64) bf16_t ldsA[2048];
  const int tid = threadIdx.x, lane = tid & 31, w = tid >> 5;
  const int n = lane & 15, half = lane >> 4;
  const int e0 = blockIdx.x * 16;

  stage_async_frag(r_frag + (size_t)blockIdx.x * 2048, ldsA, tid);
  if (blockIdx.x + 1 < (int)gridDim.x)
    __builtin_prefetch(r_frag + (size_t)(blockIdx.x + 1) * 2048 + tid * 8, 0, 1);
  wait_async0();
  __syncthreads();

#pragma unroll
  for (int tt = 0; tt < 3; ++tt) {
    int ct = w + tt * 8;
    f32x8 acc = gemm_tile(ldsA, wb + OFF_WRE, ct, lane);
    int c = ct * 16 + n;
    int h = c / 48;                 // uniform per 16-tile (16 divides 48)
    float b = wreb[c];
#pragma unroll
    for (int r = 0; r < 8; ++r) {
      int e = e0 + r + 8 * half;
      int src = ei[e], dst = ei[EE + e];
      float attn = exbuf[(size_t)e * HH + h] / den[(size_t)dst * HH + h] *
                   sqrtf(nee[e]) * INV_SQRT_F;
      float dd = d_ij[e];
      float cw = (dd < RCUT) ? 0.5f * (__cosf(dd * PI_OVER_RC) + 1.0f) : 0.0f;
      float re = acc[r] + b;
      float xe = attn * val[(size_t)src * F3 + c] +
                 re * x[(size_t)src * F3 + c] * cw;
      if (c < FF) {                               // o_s -> s_out
        atomicAdd(&s_out[(size_t)dst * FF + c], xe);
      } else if (c < 2 * FF) {                    // o_d * dir -> t_out
        int f = c - FF;
#pragma unroll
        for (int ax = 0; ax < 3; ++ax)
          atomicAdd(&t_out[((size_t)dst * 3 + ax) * FF + f],
                    xe * dirv[(size_t)e * 3 + ax]);
      } else {                                    // o_t * t[src] -> t_out
        int f = c - 2 * FF;
#pragma unroll
        for (int ax = 0; ax < 3; ++ax)
          atomicAdd(&t_out[((size_t)dst * 3 + ax) * FF + f],
                    xe * t_in[((size_t)src * 3 + ax) * FF + f]);
      }
    }
  }
}

// w1 = t_out @ vecq^T, w2 = t_out @ veck0^T over 3N rows.
__global__ void __launch_bounds__(256) wqk_kernel(
    const float* __restrict__ t_out, const bf16_t* __restrict__ wb,
    float* __restrict__ w1, float* __restrict__ w2) {
  __shared__ __align__(64) bf16_t ldsA[2048];
  const int tid = threadIdx.x, lane = tid & 31, w = tid >> 5;
  const int row0 = blockIdx.x * 16;
  stage_A(t_out, row0, ldsA, tid);
  __syncthreads();
  f32x8 a1 = gemm_tile(ldsA, wb + OFF_VECQ, w, lane);
  store_tile(w1, row0, FF, w * 16, lane, a1);
  f32x8 a2 = gemm_tile(ldsA, wb + OFF_VECK, w, lane);
  store_tile(w2, row0, FF, w * 16, lane, a2);
}

// f_out: silu(eup(r_ij)) (WMMA) * rejection dot product + residual.
__global__ void __launch_bounds__(256) edge_fout_kernel(
    const bf16_t* __restrict__ r_frag, const float* __restrict__ r_ij,
    const bf16_t* __restrict__ wb, const float* __restrict__ eupb,
    const int* __restrict__ ei, const float* __restrict__ dirv,
    const float* __restrict__ w1, const float* __restrict__ w2,
    float* __restrict__ f_out) {
  __shared__ __align__(64) bf16_t ldsA[2048];
  const int tid = threadIdx.x, lane = tid & 31, w = tid >> 5;
  const int n = lane & 15, half = lane >> 4;
  const int e0 = blockIdx.x * 16;

  stage_async_frag(r_frag + (size_t)blockIdx.x * 2048, ldsA, tid);
  if (blockIdx.x + 1 < (int)gridDim.x)
    __builtin_prefetch(r_frag + (size_t)(blockIdx.x + 1) * 2048 + tid * 8, 0, 1);
  wait_async0();
  __syncthreads();

  f32x8 acc = gemm_tile(ldsA, wb + OFF_EUP, w, lane);
  float b = eupb[w * 16 + n];
  int f = w * 16 + n;
#pragma unroll
  for (int r = 0; r < 8; ++r) {
    int e = e0 + r + 8 * half;
    int src = ei[e], dst = ei[EE + e];
    float g = silu(acc[r] + b);
    float dx = dirv[(size_t)e * 3 + 0];
    float dy = dirv[(size_t)e * 3 + 1];
    float dz = dirv[(size_t)e * 3 + 2];
    float a1x = w1[((size_t)dst * 3 + 0) * FF + f];
    float a1y = w1[((size_t)dst * 3 + 1) * FF + f];
    float a1z = w1[((size_t)dst * 3 + 2) * FF + f];
    float a2x = w2[((size_t)src * 3 + 0) * FF + f];
    float a2y = w2[((size_t)src * 3 + 1) * FF + f];
    float a2z = w2[((size_t)src * 3 + 2) * FF + f];
    float p1 = a1x * dx + a1y * dy + a1z * dz;
    float e1x = a1x - p1 * dx, e1y = a1y - p1 * dy, e1z = a1z - p1 * dz;
    float p2 = -(a2x * dx + a2y * dy + a2z * dz);  // projection onto -dir
    float e2x = a2x + p2 * dx, e2y = a2y + p2 * dy, e2z = a2z + p2 * dz;
    float wdot = e1x * e2x + e1y * e2y + e1z * e2z;
    f_out[(size_t)e * FF + f] = r_ij[(size_t)e * FF + f] + g * wdot;
  }
}

// ---------------------------------------------------------------------------
extern "C" void kernel_launch(void* const* d_in, const int* in_sizes, int n_in,
                              void* d_out, int out_size, void* d_ws, size_t ws_size,
                              hipStream_t stream) {
  (void)in_sizes; (void)n_in; (void)out_size; (void)ws_size;
  const int*   ei     = (const int*)d_in[0];
  const float* s      = (const float*)d_in[1];
  const float* t      = (const float*)d_in[2];
  const float* dirv   = (const float*)d_in[3];
  const float* r_ij   = (const float*)d_in[4];
  const float* d_ij   = (const float*)d_in[5];
  const float* nee    = (const float*)d_in[6];
  const float* qw_W   = (const float*)d_in[7];
  const float* qw_b   = (const float*)d_in[8];
  const float* kw_W   = (const float*)d_in[9];
  const float* kw_b   = (const float*)d_in[10];
  const float* gs1_W  = (const float*)d_in[11];
  const float* gs1_b  = (const float*)d_in[12];
  const float* gs2_W  = (const float*)d_in[13];
  const float* gs2_b  = (const float*)d_in[14];
  const float* gv1_W  = (const float*)d_in[15];
  const float* gv1_b  = (const float*)d_in[16];
  const float* gv2_W  = (const float*)d_in[17];
  const float* gv2_b  = (const float*)d_in[18];
  const float* phik_W = (const float*)d_in[19];
  const float* phik_b = (const float*)d_in[20];
  const float* wre_W  = (const float*)d_in[21];
  const float* wre_b  = (const float*)d_in[22];
  const float* eup_W  = (const float*)d_in[23];
  const float* eup_b  = (const float*)d_in[24];
  const float* vecq_W = (const float*)d_in[25];
  const float* veck_W = (const float*)d_in[26];

  char* wsc = (char*)d_ws;
  bf16_t* wb = (bf16_t*)wsc;
  size_t off = WB_ELEMS * sizeof(bf16_t);           // 557056, 256B aligned
  float* q     = (float*)(wsc + off); off += (size_t)NN * FF * 4;
  float* k     = (float*)(wsc + off); off += (size_t)NN * FF * 4;
  float* x     = (float*)(wsc + off); off += (size_t)NN * F3 * 4;
  float* val   = (float*)(wsc + off); off += (size_t)NN * F3 * 4;
  float* exbuf = (float*)(wsc + off); off += (size_t)EE * HH * 4;
  float* mbuf  = (float*)(wsc + off); off += (size_t)NN * HH * 4;
  float* den   = (float*)(wsc + off); off += (size_t)NN * HH * 4;
  float* w1    = (float*)(wsc + off); off += (size_t)3 * NN * FF * 4;
  float* w2    = (float*)(wsc + off); off += (size_t)3 * NN * FF * 4;
  bf16_t* r_frag = (bf16_t*)(wsc + off); off += (size_t)EE * FF * 2;

  float* s_out = (float*)d_out;
  float* t_out = s_out + (size_t)NN * FF;
  float* f_out = t_out + (size_t)NN * F3;

  auto sw = [&](const float* W, size_t o, int cout) {
    int total = cout * FF;
    swizzle_w_kernel<<<(total + 255) / 256, 256, 0, stream>>>(W, wb + o, total);
  };
  sw(qw_W,   OFF_QW,   FF);
  sw(kw_W,   OFF_KW,   FF);
  sw(gs1_W,  OFF_GS1,  FF);
  sw(gs2_W,  OFF_GS2,  F3);
  sw(gv1_W,  OFF_GV1,  FF);
  sw(gv2_W,  OFF_GV2,  F3);
  sw(phik_W, OFF_PHIK, FF);
  sw(wre_W,  OFF_WRE,  F3);
  sw(eup_W,  OFF_EUP,  FF);
  sw(vecq_W, OFF_VECQ, FF);
  sw(veck_W, OFF_VECK, FF);

  swizzle_r_kernel<<<((size_t)EE * FF + 255) / 256, 256, 0, stream>>>(r_ij, r_frag);

  init_kernel<<<((size_t)NN * F3 + 255) / 256, 256, 0, stream>>>(
      s, t, s_out, t_out, mbuf, den);

  node_gemm_kernel<<<NN / 16, 256, 0, stream>>>(
      s, wb, qw_b, kw_b, gs1_b, gs2_b, gv1_b, gv2_b, q, k, x, val);

  edge_logits_kernel<<<EE / 16, 256, 0, stream>>>(
      r_frag, wb, phik_b, ei, q, k, exbuf, mbuf);

  softmax_ex_kernel<<<((size_t)EE * HH + 255) / 256, 256, 0, stream>>>(
      exbuf, mbuf, den, ei);

  edge_main_kernel<<<EE / 16, 256, 0, stream>>>(
      r_frag, wb, wre_b, ei, exbuf, den, nee, d_ij, dirv, x, val, t,
      s_out, t_out);

  wqk_kernel<<<(3 * NN) / 16, 256, 0, stream>>>(t_out, wb, w1, w2);

  edge_fout_kernel<<<EE / 16, 256, 0, stream>>>(
      r_frag, r_ij, wb, eup_b, ei, dirv, w1, w2, f_out);
}
